// GCNEncoder_89859305767430
// MI455X (gfx1250) — matrix operand order repaired and checked by reading, not compiled, not run on hardware
//
#include <hip/hip_runtime.h>
#include <hip/hip_bf16.h>
#include <stdint.h>

typedef __attribute__((ext_vector_type(16))) __bf16 v16bf;
typedef __attribute__((ext_vector_type(8)))  float  v8f;

// Hardware bf16 convert (RNE) — lowers to v_cvt_*bf16_f32 on gfx1250.
__device__ __forceinline__ __bf16 f2bf(float f) { return (__bf16)f; }

__device__ __forceinline__ unsigned int packbf(float a, float b) {
    __bf16 x = (__bf16)a, y = (__bf16)b;
    unsigned short lo, hi;
    __builtin_memcpy(&lo, &x, 2);
    __builtin_memcpy(&hi, &y, 2);
    return (unsigned int)lo | ((unsigned int)hi << 16);
}

// ---------------------------------------------------------------------------
// H[M x 256] = X[M x 256] @ W[256 x 256], bf16 WMMA, f32 accumulate.
// Block: 256 threads = 8 waves; block tile = 32 rows x 256 cols.
// Wave (msub, ngrp): 16 rows x 64 cols = 4 accumulators of 16x16.
// Requires M % 32 == 0 (holds: M = 100000).
// ---------------------------------------------------------------------------
__global__ __launch_bounds__(256) void gcn_gemm_bf16(
    const float* __restrict__ X, const float* __restrict__ W,
    float* __restrict__ H, int M)
{
    // W^T slice for K in [k, k+32): wt[n][kk/2] packs bf16(K=2j | K=2j+1 << 16)
    // row stride 20 dwords = 80B -> 16B aligned fragment reads, conflict-free.
    __shared__ unsigned int wt[256][20];

    const int tid  = threadIdx.x;
    const int lane = tid & 31;
    const int wave = tid >> 5;
    const int msub = wave & 1;      // which 16-row half of the 32-row block tile
    const int ngrp = wave >> 1;     // which 64-col group
    const int lo16 = lane & 15;
    const int hi   = lane >> 4;     // 0: lanes 0-15, 1: lanes 16-31

    const int rowbase = blockIdx.x * 32 + msub * 16;
    const float* __restrict__ xr = X + (size_t)(rowbase + lo16) * 256;

    v8f acc[4] = {};

    for (int k = 0; k < 256; k += 32) {
        // ---- stage W[k..k+31][:] transposed into LDS as bf16 pairs ----
        // thread t owns output column n = t; coalesced global reads per kk.
        #pragma unroll
        for (int kk = 0; kk < 32; kk += 2) {
            float w0 = W[(k + kk)     * 256 + tid];
            float w1 = W[(k + kk + 1) * 256 + tid];
            wt[tid][kk >> 1] = packbf(w0, w1);
        }
        __syncthreads();

        // ---- A fragment: contiguous k order (k = 16*hi + s), same as B ----
        // lanes 0-15:  row=lane,    elems 0-15 = K k..k+15
        // lanes 16-31: row=lane-16, elems 0-15 = K k+16..k+31
        const int base = k + (hi << 4);
        float4 a0 = *(const float4*)(xr + base);
        float4 a1 = *(const float4*)(xr + base + 4);
        float4 a2 = *(const float4*)(xr + base + 8);
        float4 a3 = *(const float4*)(xr + base + 12);

        // ---- issue all 4 B fragments first (one dscnt wait, 4 WMMAs b2b) ----
        // B in the same contiguous k order: lanes 0-15 col n K k..k+15; lanes 16-31 K k+16..k+31
        union BU { uint4 u[2]; v16bf v; };
        BU bu[4];
        #pragma unroll
        for (int j = 0; j < 4; ++j) {
            const unsigned int* rp = &wt[ngrp * 64 + j * 16 + lo16][hi << 3];
            bu[j].u[0] = *(const uint4*)rp;
            bu[j].u[1] = *(const uint4*)(rp + 4);
        }

        v16bf af;
        af[0]  = f2bf(a0.x); af[1]  = f2bf(a0.y); af[2]  = f2bf(a0.z); af[3]  = f2bf(a0.w);
        af[4]  = f2bf(a1.x); af[5]  = f2bf(a1.y); af[6]  = f2bf(a1.z); af[7]  = f2bf(a1.w);
        af[8]  = f2bf(a2.x); af[9]  = f2bf(a2.y); af[10] = f2bf(a2.z); af[11] = f2bf(a2.w);
        af[12] = f2bf(a3.x); af[13] = f2bf(a3.y); af[14] = f2bf(a3.z); af[15] = f2bf(a3.w);

        #pragma unroll
        for (int j = 0; j < 4; ++j) {
            acc[j] = __builtin_amdgcn_wmma_f32_16x16x32_bf16(
                false, af, false, bu[j].v, (short)0, acc[j], false, false);
        }
        __syncthreads();
    }

    // ---- store C: vgpr r -> row rowbase + hi*8 + r, col = n0 + lo16 ----
    const int srow = rowbase + (hi << 3);
    #pragma unroll
    for (int j = 0; j < 4; ++j) {
        float* hp = H + (size_t)srow * 256 + (ngrp * 64 + j * 16 + lo16);
        #pragma unroll
        for (int r = 0; r < 8; ++r)
            hp[(size_t)r * 256] = acc[j][r];
    }
}

// ---------------------------------------------------------------------------
// degree count / rsqrt finalize
// ---------------------------------------------------------------------------
__global__ void deg_count(const int* __restrict__ dst, int E, float* __restrict__ deg)
{
    int i = blockIdx.x * blockDim.x + threadIdx.x;
    if (i < E) atomicAdd(&deg[dst[i]], 1.0f);
}

__global__ void deg_finalize(float* __restrict__ deg, int N)
{
    int i = blockIdx.x * blockDim.x + threadIdx.x;
    if (i < N) deg[i] = rsqrtf(deg[i] + 1.0f);  // +1 = self loop, always > 0
}

// ---------------------------------------------------------------------------
// edge scatter: one wave per edge; lane covers 8 channels (2x float4 gather,
// 8x global_atomic_add_f32 scatter). Coalesced 512B loads across the wave.
// ---------------------------------------------------------------------------
__global__ __launch_bounds__(256) void aggregate(
    const float* __restrict__ h, const int* __restrict__ src,
    const int* __restrict__ dst, const float* __restrict__ dis,
    float* __restrict__ agg, int E)
{
    const int lane = threadIdx.x & 31;
    const int e = (int)((blockIdx.x * blockDim.x + threadIdx.x) >> 5);
    if (e >= E) return;
    const int s = src[e];
    const int d = dst[e];
    const float norm = dis[s] * dis[d];

    const float4* hp = (const float4*)(h + (size_t)s * 256);
    float4 v0 = hp[lane];        // channels lane*4 .. +3
    float4 v1 = hp[lane + 32];   // channels 128 + lane*4 .. +3

    float* ap = agg + (size_t)d * 256;
    const int c0 = lane * 4, c1 = 128 + lane * 4;
    atomicAdd(&ap[c0 + 0], v0.x * norm);
    atomicAdd(&ap[c0 + 1], v0.y * norm);
    atomicAdd(&ap[c0 + 2], v0.z * norm);
    atomicAdd(&ap[c0 + 3], v0.w * norm);
    atomicAdd(&ap[c1 + 0], v1.x * norm);
    atomicAdd(&ap[c1 + 1], v1.y * norm);
    atomicAdd(&ap[c1 + 2], v1.z * norm);
    atomicAdd(&ap[c1 + 3], v1.w * norm);
}

// ---------------------------------------------------------------------------
// out = [relu](agg + h*dis^2 (self loop) + bias), in place on agg. float4 lanes.
// ---------------------------------------------------------------------------
__global__ __launch_bounds__(256) void finalize(
    float* __restrict__ agg, const float* __restrict__ h,
    const float* __restrict__ dis, const float* __restrict__ bias,
    int N, int relu)
{
    size_t idx = (size_t)blockIdx.x * blockDim.x + threadIdx.x;
    if (idx >= (size_t)N * 64) return;
    const int node = (int)(idx >> 6);
    const int q = (int)(idx & 63);
    const float ds = dis[node];
    const float dsq = ds * ds;
    float4 a  = ((const float4*)agg)[idx];
    float4 hh = ((const float4*)h)[idx];
    float4 bb = ((const float4*)bias)[q];
    float4 r;
    r.x = fmaf(hh.x, dsq, a.x) + bb.x;
    r.y = fmaf(hh.y, dsq, a.y) + bb.y;
    r.z = fmaf(hh.z, dsq, a.z) + bb.z;
    r.w = fmaf(hh.w, dsq, a.w) + bb.w;
    if (relu) {
        r.x = fmaxf(r.x, 0.0f); r.y = fmaxf(r.y, 0.0f);
        r.z = fmaxf(r.z, 0.0f); r.w = fmaxf(r.w, 0.0f);
    }
    ((float4*)agg)[idx] = r;
}

// ---------------------------------------------------------------------------
// column mean: thread t owns channel t; block sums a row range, one atomic.
// ---------------------------------------------------------------------------
__global__ __launch_bounds__(256) void mean_reduce(
    const float* __restrict__ buf, float* __restrict__ out, int N, int rows)
{
    const int c = threadIdx.x;
    const int i0 = blockIdx.x * rows;
    int i1 = i0 + rows; if (i1 > N) i1 = N;
    float s = 0.0f;
    for (int i = i0; i < i1; ++i) s += buf[(size_t)i * 256 + c];
    atomicAdd(&out[c], s * (1.0f / (float)N));
}

// ---------------------------------------------------------------------------
extern "C" void kernel_launch(void* const* d_in, const int* in_sizes, int n_in,
                              void* d_out, int out_size, void* d_ws, size_t ws_size,
                              hipStream_t stream)
{
    const float* x  = (const float*)d_in[0];
    const float* W1 = (const float*)d_in[1];
    const float* b1 = (const float*)d_in[2];
    const float* W2 = (const float*)d_in[3];
    const float* b2 = (const float*)d_in[4];
    const int*   ei = (const int*)d_in[5];

    const int N = in_sizes[0] / 256;
    const int E = in_sizes[5] / 2;
    const int* src = ei;
    const int* dst = ei + E;

    float* dis  = (float*)d_ws;                          // N (deg -> rsqrt(deg))
    float* bufA = dis + (((size_t)N + 255) & ~(size_t)255);
    float* bufB = bufA + (size_t)N * 256;

    // degrees
    hipMemsetAsync(dis, 0, (size_t)N * sizeof(float), stream);
    deg_count<<<(E + 255) / 256, 256, 0, stream>>>(dst, E, dis);
    deg_finalize<<<(N + 255) / 256, 256, 0, stream>>>(dis, N);

    const int mBlocks = N / 32;                          // N % 32 == 0
    const int fBlocks = (int)(((size_t)N * 64 + 255) / 256);
    const int eBlocks = (E + 7) / 8;

    // layer 1
    gcn_gemm_bf16<<<mBlocks, 256, 0, stream>>>(x, W1, bufA, N);
    hipMemsetAsync(bufB, 0, (size_t)N * 256 * sizeof(float), stream);
    aggregate<<<eBlocks, 256, 0, stream>>>(bufA, src, dst, dis, bufB, E);
    finalize<<<fBlocks, 256, 0, stream>>>(bufB, bufA, dis, b1, N, 1);

    // layer 2
    gcn_gemm_bf16<<<mBlocks, 256, 0, stream>>>(bufB, W2, bufA, N);
    hipMemsetAsync(bufB, 0, (size_t)N * 256 * sizeof(float), stream);
    aggregate<<<eBlocks, 256, 0, stream>>>(bufA, src, dst, dis, bufB, E);
    finalize<<<fBlocks, 256, 0, stream>>>(bufB, bufA, dis, b2, N, 0);

    // mean over nodes -> [1, 256]
    hipMemsetAsync(d_out, 0, (size_t)out_size * sizeof(float), stream);
    const int mrBlocks = 400;
    const int rows = (N + mrBlocks - 1) / mrBlocks;
    mean_reduce<<<mrBlocks, 256, 0, stream>>>(bufB, (float*)d_out, N, rows);
}
